// CwRNNCell_90692529422438
// MI455X (gfx1250) — matrix operand order repaired and checked by reading, not compiled, not run on hardware
//
#include <hip/hip_runtime.h>

// ---------------------------------------------------------------------------
// Clockwork RNN for MI455X (gfx1250)
//   Phase 1: preX[t][i] = x[t]·W_ih[i] + b_ih[i] + b_hh[i]   (bf16 WMMA GEMM)
//   Phase 2: sequential clockwork recurrence (persistent kernel, fp32 VALU,
//            module-sparse: module j only computes when t % 2^j == 0)
// ---------------------------------------------------------------------------

#define T_STEPS   4096
#define IN_SIZE   1024
#define HIDDEN    2048
#define MOD_SIZE  256
#define N_MOD     8

typedef __attribute__((ext_vector_type(16))) __bf16 v16bf;
typedef __attribute__((ext_vector_type(8)))  __bf16 v8bf;
typedef __attribute__((ext_vector_type(4)))  __bf16 v4bf;
typedef __attribute__((ext_vector_type(8)))  float  v8f;

// ---- fp32 -> bf16 (round to nearest even) ---------------------------------
__device__ inline __bf16 f2bf(float f) {
    union { float f; unsigned u; } v; v.f = f;
    unsigned r = v.u + 0x7FFFu + ((v.u >> 16) & 1u);
    unsigned short h = (unsigned short)(r >> 16);
    union { unsigned short s; __bf16 b; } o; o.s = h;
    return o.b;
}

__device__ inline void store4bf(__bf16* p, float4 v) {
    v4bf o;
    o[0] = f2bf(v.x); o[1] = f2bf(v.y); o[2] = f2bf(v.z); o[3] = f2bf(v.w);
    *(v4bf*)p = o;
}

// ---- WMMA fragment gathers from LDS (CDNA5 16-bit layouts, wave32) --------
// A (16x32, MxK): lane l<16 holds row M=l, K chunks {0..7}u{16..23};
//                 lane l+16 same row, chunks {8..15}u{24..31}. sel = l>>4.
__device__ inline v16bf ldsA_frag(const __bf16* rowbase, int sel) {
    v8bf a0 = *(const v8bf*)(rowbase + sel * 8);
    v8bf a1 = *(const v8bf*)(rowbase + sel * 8 + 16);
    v16bf r;
#pragma unroll
    for (int i = 0; i < 8; ++i) { r[i] = a0[i]; r[i + 8] = a1[i]; }
    return r;
}
// B (32x16, KxN): lane l<16 holds col N=l, K chunks {0..7}u{16..23}; lane l+16 {8..15}u{24..31}.
__device__ inline v16bf ldsB_frag(const __bf16* rowbase, int sel) {
    v8bf b0 = *(const v8bf*)(rowbase + sel * 8);
    v8bf b1 = *(const v8bf*)(rowbase + sel * 8 + 16);
    v16bf r;
#pragma unroll
    for (int i = 0; i < 8; ++i) { r[i] = b0[i]; r[i + 8] = b1[i]; }
    return r;
}

// ---------------------------------------------------------------------------
// Phase 1: bf16 WMMA GEMM.  Block tile 128(M) x 64(N), K-step 32.
// 256 threads = 8 waves, arranged 4(M) x 2(N); each wave owns a 32x32 subtile
// = 2x2 WMMA fragments (v_wmma_f32_16x16x32_bf16, fp32 accumulate).
// ---------------------------------------------------------------------------
#define TM 128
#define TN 64
#define LSTR 56   // LDS row stride in bf16 elems: 112B, 16B-aligned, spreads banks

__global__ __launch_bounds__(256)
void cwrnn_gemm(const float* __restrict__ X, const float* __restrict__ Wih,
                const float* __restrict__ b_ih, const float* __restrict__ b_hh,
                float* __restrict__ preX) {
    __shared__ __align__(16) __bf16 lA[TM * LSTR];
    __shared__ __align__(16) __bf16 lB[TN * LSTR];

    const int m0 = blockIdx.x * TM;
    const int n0 = blockIdx.y * TN;
    const int tid  = threadIdx.x;
    const int wave = tid >> 5;
    const int lane = tid & 31;
    const int wm = wave & 3;        // 0..3 -> M sub-tile
    const int wn = wave >> 2;       // 0..1 -> N sub-tile
    const int fr  = lane & 15;
    const int sel = lane >> 4;

    v8f acc[2][2] = {};

    for (int kk = 0; kk < IN_SIZE; kk += 32) {
        // ---- stage A tile (128x32 fp32 -> bf16) ----
#pragma unroll
        for (int i = 0; i < 4; ++i) {
            int c = tid + 256 * i;          // 1024 float4 chunks
            int r = c >> 3, c4 = c & 7;
            float4 v = *(const float4*)(X + (size_t)(m0 + r) * IN_SIZE + kk + c4 * 4);
            store4bf(&lA[r * LSTR + c4 * 4], v);
        }
        // ---- stage B tile (64x32 fp32 -> bf16): rows of W_ih ----
#pragma unroll
        for (int i = 0; i < 2; ++i) {
            int c = tid + 256 * i;          // 512 float4 chunks
            int r = c >> 3, c4 = c & 7;
            float4 v = *(const float4*)(Wih + (size_t)(n0 + r) * IN_SIZE + kk + c4 * 4);
            store4bf(&lB[r * LSTR + c4 * 4], v);
        }
        __syncthreads();

#pragma unroll
        for (int mi = 0; mi < 2; ++mi) {
            v16bf a = ldsA_frag(&lA[(wm * 32 + mi * 16 + fr) * LSTR], sel);
#pragma unroll
            for (int ni = 0; ni < 2; ++ni) {
                v16bf b = ldsB_frag(&lB[(wn * 32 + ni * 16 + fr) * LSTR], sel);
                acc[mi][ni] = __builtin_amdgcn_wmma_f32_16x16x32_bf16(
                    false, a, false, b, (short)0, acc[mi][ni], false, false);
            }
        }
        __syncthreads();
    }

    // ---- epilogue: D layout — VGPR r, lane l -> (M = r + 8*(l>>4), N = l&15)
#pragma unroll
    for (int mi = 0; mi < 2; ++mi) {
#pragma unroll
        for (int ni = 0; ni < 2; ++ni) {
            int n = n0 + wn * 32 + ni * 16 + fr;
            float bias = b_ih[n] + b_hh[n];
            int mb = m0 + wm * 32 + mi * 16 + 8 * sel;
#pragma unroll
            for (int r = 0; r < 8; ++r)
                preX[(size_t)(mb + r) * HIDDEN + n] = acc[mi][ni][r] + bias;
        }
    }
}

// ---------------------------------------------------------------------------
// Phase 2: persistent clockwork recurrence.
// 8 workgroups (module g), 1024 threads: 4 lanes per output row (K quarters),
// LDS reduction, tanh.  Cross-WG split barrier: monotonic global atomic.
// ---------------------------------------------------------------------------
__device__ inline void grid_barrier(unsigned* bar, unsigned target) {
    __syncthreads();
    if (threadIdx.x == 0) {
        __threadfence();                       // publish writes
        atomicAdd(bar, 1u);
        while (atomicAdd(bar, 0u) < target)    // atomic load
            __builtin_amdgcn_s_sleep(2);
        __threadfence();                       // acquire
    }
    __syncthreads();
}

__global__ __launch_bounds__(1024)
void cwrnn_rec(const float* __restrict__ preX, const float* __restrict__ Whh,
               float* __restrict__ h, unsigned* __restrict__ bar,
               float* __restrict__ out) {
    __shared__ float hs[HIDDEN];
    __shared__ float part[1024];

    const int g   = blockIdx.x;            // module id, period = 1<<g
    const int tid = threadIdx.x;
    const int rl  = tid >> 2;              // local row 0..255
    const int q   = tid & 3;               // K quarter
    const int row = g * MOD_SIZE + rl;
    const float* wrow = Whh + (size_t)row * HIDDEN + q * 512;

    __builtin_prefetch(wrow, 0, 3);        // global_prefetch_b8: warm L2/near
    unsigned done = 0;

    for (int t = 1; t <= T_STEPS; ++t) {
        const bool active = ((t & ((1 << g) - 1)) == 0);

        if (active) {                      // snapshot h_{t-1} into LDS
            hs[tid]        = h[tid];
            hs[tid + 1024] = h[tid + 1024];
        }
        done += N_MOD;                     // barrier A: all snapshots done
        grid_barrier(bar, done);

        if (active) {
            float s = 0.f;
            const float4* w4 = (const float4*)wrow;
            const float*  hp = &hs[q * 512];
#pragma unroll 8
            for (int i = 0; i < 128; ++i) {
                float4 wv = w4[i];
                s += wv.x * hp[i * 4 + 0] + wv.y * hp[i * 4 + 1]
                   + wv.z * hp[i * 4 + 2] + wv.w * hp[i * 4 + 3];
            }
            part[tid] = s;
        }
        __syncthreads();
        if (active && q == 0) {
            float pre = part[tid] + part[tid + 1] + part[tid + 2] + part[tid + 3]
                      + preX[(size_t)(t - 1) * HIDDEN + row];
            h[row] = tanhf(pre);           // inactive rows keep old value
        }
        done += N_MOD;                     // barrier B: h_t published
        grid_barrier(bar, done);

        if (tid < MOD_SIZE)                // each WG emits its own block of h_t
            out[(size_t)(t - 1) * HIDDEN + g * MOD_SIZE + tid] =
                h[g * MOD_SIZE + tid];
    }
}

// ---------------------------------------------------------------------------
__global__ void cwrnn_init(float* h, unsigned* bar) {
    int i = blockIdx.x * blockDim.x + threadIdx.x;
    if (i < HIDDEN) h[i] = 0.f;
    if (i == 0) *bar = 0u;
}

extern "C" void kernel_launch(void* const* d_in, const int* in_sizes, int n_in,
                              void* d_out, int out_size, void* d_ws, size_t ws_size,
                              hipStream_t stream) {
    const float* X    = (const float*)d_in[0];
    const float* Wih  = (const float*)d_in[1];
    const float* Whh  = (const float*)d_in[2];
    const float* b_ih = (const float*)d_in[3];
    const float* b_hh = (const float*)d_in[4];
    float* out = (float*)d_out;

    float*    ws   = (float*)d_ws;
    float*    preX = ws;                                   // 4096*2048 fp32 = 32 MB
    float*    h    = ws + (size_t)T_STEPS * HIDDEN;        // 2048 fp32
    unsigned* bar  = (unsigned*)(h + HIDDEN);              // 1 u32

    cwrnn_init<<<(HIDDEN + 255) / 256, 256, 0, stream>>>(h, bar);

    dim3 grid(T_STEPS / TM, HIDDEN / TN);                  // 32 x 32 blocks
    cwrnn_gemm<<<grid, 256, 0, stream>>>(X, Wih, b_ih, b_hh, preX);

    cwrnn_rec<<<N_MOD, 1024, 0, stream>>>(preX, Whh, h, bar, out);
}
